// LinearAttention_21835613733531
// MI455X (gfx1250) — matrix operand order repaired and checked by reading, not compiled, not run on hardware
//
#include <hip/hip_runtime.h>
#include <hip/hip_bf16.h>

// ---------------------------------------------------------------------------
// Linear attention (ELU+1 feature map) for MI455X / gfx1250.
// GEMMs: v_wmma_f32_16x16x32_bf16 fed from double-buffered LDS.
//  - A panels staged with global_load_async_to_lds_b128  (ASYNCcnt)
//  - B (weight) panels staged with TDM tensor_load_to_lds (TENSORcnt)
// fp32 accumulation throughout.
// ---------------------------------------------------------------------------

typedef __attribute__((ext_vector_type(16))) __bf16 v16bf;
typedef __attribute__((ext_vector_type(8)))  float  v8f;
typedef __attribute__((ext_vector_type(4)))  unsigned int v4u;
typedef __attribute__((ext_vector_type(8)))  int    v8i;
typedef __attribute__((ext_vector_type(4)))  int    v4i;

#define D_MODEL 1024
#define N_SEQ   8192
#define BATCH   2
#define NHEADS  16
#define DK      64
#define MTOT    (BATCH * N_SEQ)    // 16384 rows (batch folded into M)
#define NHH     (BATCH * NHEADS)   // 32 (b,h) pairs
#define KC      64                 // K-chunk staged in LDS per buffer

union TileU { v16bf v; uint4 u[2]; };

// A-matrix 16x32 bf16 tile per ISA layout:
// lane<16: row=lane, K = {k0..k0+7, k0+16..k0+23}; lane>=16: K offsets +8.
__device__ __forceinline__ v16bf load_a_tile(const __bf16* p, int row, int ld,
                                             int k0, int lane) {
    const __bf16* q = p + (size_t)row * ld + k0 + ((lane >> 4) << 3);
    TileU t;
    t.u[0] = *(const uint4*)(q);
    t.u[1] = *(const uint4*)(q + 16);
    return t.v;
}

// B-matrix 32x16 bf16 tile per ISA layout (column per lane):
// lane<16: col=lane, K = {k0..k0+7, k0+16..k0+23}; lane>=16: K offsets +8.
// "row" = row of the K-major source holding that column contiguously.
__device__ __forceinline__ v16bf load_b_tile(const __bf16* p, int row, int ld,
                                             int k0, int lane) {
    const __bf16* q = p + (size_t)row * ld + k0 + ((lane >> 4) << 3);
    TileU t;
    t.u[0] = *(const uint4*)(q);
    t.u[1] = *(const uint4*)(q + 16);
    return t.v;
}

__device__ __forceinline__ v8f wmma_bf16(v16bf a, v16bf b, v8f c) {
    return __builtin_amdgcn_wmma_f32_16x16x32_bf16(false, a, false, b,
                                                   (short)0, c, false, false);
}

// Async global->LDS 16B copy (ASYNCcnt path). LDS offset = low 32 bits of
// the generic address of a __shared__ object.
__device__ __forceinline__ void async_copy_b128(void* lds_ptr, const void* g) {
    uint32_t l = (uint32_t)(uintptr_t)lds_ptr;
    asm volatile("global_load_async_to_lds_b128 %0, %1, off"
                 :: "v"(l), "v"(g) : "memory");
}
__device__ __forceinline__ void wait_asynccnt0() {
    asm volatile("s_wait_asynccnt 0x0" ::: "memory");
}

// TDM: stage a rows x cols (elements, 2B each) tile of a row-major tensor
// [tdim1 x tdim0] into LDS. Descriptor per CDNA5 ISA.
// 2D tile -> groups 2/3 zero. Issued by one wave; tracked by TENSORcnt.
#define HAVE_TDM __has_builtin(__builtin_amdgcn_tensor_load_to_lds)
__device__ __forceinline__ void tdm_load_2d(void* lds_ptr, const void* gptr,
                                            int tdim0, int tdim1,
                                            int cols, int rows) {
#if HAVE_TDM
    const uint32_t lds = (uint32_t)(uintptr_t)lds_ptr;
    const uint64_t ga  = (uint64_t)(uintptr_t)gptr;
    v4u g0;
    g0[0] = 1u;                                  // count=1, user mode
    g0[1] = lds;                                 // lds_addr
    g0[2] = (uint32_t)ga;                        // global_addr[31:0]
    g0[3] = (uint32_t)((ga >> 32) & 0x1FFFFFFu)  // global_addr[56:32]
            | (2u << 30);                        // type=2 ("image")
    v8i g1;
    g1[0] = (1 << 16);                           // data_size=1 -> 2 bytes
    g1[1] = (int)(((unsigned)tdim0 & 0xFFFFu) << 16);        // tensor_dim0 lo
    g1[2] = (int)((((unsigned)tdim0 >> 16) & 0xFFFFu)
            | (((unsigned)tdim1 & 0xFFFFu) << 16));          // dim0 hi|dim1 lo
    g1[3] = (int)((((unsigned)tdim1 >> 16) & 0xFFFFu)
            | ((unsigned)cols << 16));                       // dim1 hi|tile_d0
    g1[4] = rows;                                // tile_dim1 (tile_dim2=0)
    g1[5] = tdim0;                               // tensor_dim0_stride[31:0]
    g1[6] = 0;
    g1[7] = 0;
    v4i z4 = {0, 0, 0, 0};
#if __clang_major__ >= 23
    v8i z8 = {0, 0, 0, 0, 0, 0, 0, 0};
    __builtin_amdgcn_tensor_load_to_lds(g0, g1, z4, z4, z8, 0);
#else
    __builtin_amdgcn_tensor_load_to_lds(g0, g1, z4, z4, 0);
#endif
#endif
}

__device__ __forceinline__ float apply_act(float y, int act) {
    return act ? ((y > 0.0f) ? (y + 1.0f) : __expf(y)) : y;  // elu(x)+1
}

__device__ __forceinline__ void store_tr(__bf16* out_tr, int row, int col,
                                         float y) {
    const int bb = row >> 13;             // row / 8192
    const int ns = row & (N_SEQ - 1);
    const int h  = col >> 6;              // col / 64
    const int d  = col & (DK - 1);
    out_tr[((((size_t)bb * NHEADS + h) * DK + d) << 13) + ns] = (__bf16)y;
}

// ---------------------------------------------------------------------------
// fp32 -> bf16 conversion (grid-stride)
// ---------------------------------------------------------------------------
__global__ void cvt_kernel(const float* __restrict__ x, __bf16* __restrict__ y,
                           int n) {
    int i = blockIdx.x * blockDim.x + threadIdx.x;
    int stride = gridDim.x * blockDim.x;
    for (; i < n; i += stride) y[i] = (__bf16)x[i];
}

// ---------------------------------------------------------------------------
// Tiled GEMM: OUT = act(A @ W^T + bias).
// A: [M,K] bf16 row-major; W: [N,K] bf16 row-major (W^T columns = W rows).
// Block = 8 waves -> 64(M) x 128(N) tile; each wave owns a 32x32 register
// tile (2x2 WMMA tiles, 4 accumulators). K consumed in KC=64 chunks,
// double-buffered in LDS: A via per-lane async copies, B via one TDM
// descriptor per chunk. Fragments read via ds_load_b128.
// ---------------------------------------------------------------------------
__global__ void __launch_bounds__(256) gemm_tile_kernel(
    const __bf16* __restrict__ A, const __bf16* __restrict__ W,
    const float* __restrict__ bias,
    __bf16* __restrict__ out_rm, __bf16* __restrict__ out_tr,
    float* __restrict__ out_f32,
    int M, int N, int K, int act)
{
    __shared__ __align__(16) __bf16 sA[2][64 * KC];    // 2 x 8KB
    __shared__ __align__(16) __bf16 sB[2][128 * KC];   // 2 x 16KB

    const int tid  = threadIdx.x;
    const int lane = tid & 31;
    const int wave = tid >> 5;
    const int mblk = blockIdx.y * 64;
    const int nblk = blockIdx.x * 128;
    const int mt = (wave >> 2) * 32;   // wave M offset inside block tile
    const int nt = (wave & 3) * 32;    // wave N offset inside block tile

    // Stage one K-chunk: A panel 64xKC async (512 x 16B, 2/thread),
    // B panel 128xKC via TDM (wave 0 issues the descriptor).
    auto stage = [&](int buf, int kc) {
        #pragma unroll
        for (int i = 0; i < 2; ++i) {
            const int lin = i * 256 + tid;
            const int row = lin >> 3, c = lin & 7;
            async_copy_b128(&sA[buf][row * KC + c * 8],
                            A + (size_t)(mblk + row) * K + kc + c * 8);
        }
#if HAVE_TDM
        if (wave == 0) {
            tdm_load_2d(&sB[buf][0], W + (size_t)nblk * K + kc,
                        K, N, KC, 128);
        }
#else
        #pragma unroll
        for (int i = 0; i < 4; ++i) {
            const int lin = i * 256 + tid;
            const int row = lin >> 3, c = lin & 7;
            async_copy_b128(&sB[buf][row * KC + c * 8],
                            W + (size_t)(nblk + row) * K + kc + c * 8);
        }
#endif
    };

    v8f acc00 = {}, acc01 = {}, acc10 = {}, acc11 = {};

    stage(0, 0);
    wait_asynccnt0();
    __builtin_amdgcn_s_wait_tensorcnt(0);
    __syncthreads();

    const int nchunks = K / KC;
    for (int ch = 0; ch < nchunks; ++ch) {
        const int cur = ch & 1;
        if (ch + 1 < nchunks) stage(cur ^ 1, (ch + 1) * KC);

        const __bf16* pa = sA[cur];
        const __bf16* pb = sB[cur];
        #pragma unroll
        for (int k0 = 0; k0 < KC; k0 += 32) {
            v16bf a0 = load_a_tile(pa, mt + (lane & 15), KC, k0, lane);
            v16bf a1 = load_a_tile(pa, mt + 16 + (lane & 15), KC, k0, lane);
            v16bf b0 = load_b_tile(pb, nt + (lane & 15), KC, k0, lane);
            v16bf b1 = load_b_tile(pb, nt + 16 + (lane & 15), KC, k0, lane);
            acc00 = wmma_bf16(a0, b0, acc00);
            acc01 = wmma_bf16(a0, b1, acc01);
            acc10 = wmma_bf16(a1, b0, acc10);
            acc11 = wmma_bf16(a1, b1, acc11);
        }
        wait_asynccnt0();
        __builtin_amdgcn_s_wait_tensorcnt(0);
        __syncthreads();
    }

    // Epilogue: D layout lane<16: M=r, lane>=16: M=r+8; col = lane&15.
    const int col0 = nblk + nt + (lane & 15);
    const int col1 = col0 + 16;
    const float bv0 = bias[col0];
    const float bv1 = bias[col1];
    #pragma unroll
    for (int half = 0; half < 2; ++half) {
        const v8f aL = half ? acc10 : acc00;
        const v8f aR = half ? acc11 : acc01;
        const int rbase = mblk + mt + half * 16 + ((lane >> 4) << 3);
        #pragma unroll
        for (int r = 0; r < 8; ++r) {
            const float y0 = apply_act(aL[r] + bv0, act);
            const float y1 = apply_act(aR[r] + bv1, act);
            const int row = rbase + r;
            if (out_f32) {
                out_f32[(size_t)row * N + col0] = y0;
                out_f32[(size_t)row * N + col1] = y1;
            }
            if (out_rm) {
                out_rm[(size_t)row * N + col0] = (__bf16)y0;
                out_rm[(size_t)row * N + col1] = (__bf16)y1;
            }
            if (out_tr) {
                store_tr(out_tr, row, col0, y0);
                store_tr(out_tr, row, col1, y1);
            }
        }
    }
}

// ---------------------------------------------------------------------------
// kvT[m][d] = sum_n v[n][m] * k[n][d], per (b,h), from the head-major
// transposed copies (both WMMA operands contiguous along n).
// 16 waves = 4x4 grid of 16x16 tiles over the 64x64 output; 256 K-steps.
// ---------------------------------------------------------------------------
__global__ void __launch_bounds__(512) kv_kernel(
    const __bf16* __restrict__ kfT, const __bf16* __restrict__ vfT,
    __bf16* __restrict__ kvT)
{
    const int hh = blockIdx.x;          // (b,h) pair 0..31
    const int lane = threadIdx.x & 31;
    const int wave = threadIdx.x >> 5;  // 0..15
    const int mtile = wave >> 2, dtile = wave & 3;
    const __bf16* kh = kfT + (size_t)hh * DK * N_SEQ;
    const __bf16* vh = vfT + (size_t)hh * DK * N_SEQ;
    const int arow = mtile * 16 + (lane & 15);   // v^T row (m)
    const int brow = dtile * 16 + (lane & 15);   // k column (d)

    v8f acc = {};
    for (int n0 = 0; n0 < N_SEQ; n0 += 32) {
        __builtin_prefetch(vh + (size_t)arow * N_SEQ + n0 + 256, 0, 0);
        v16bf a = load_a_tile(vh, arow, N_SEQ, n0, lane);
        v16bf b = load_b_tile(kh, brow, N_SEQ, n0, lane);
        acc = wmma_bf16(a, b, acc);
    }

    const int d = dtile * 16 + (lane & 15);
    const int rbase = mtile * 16 + ((lane >> 4) << 3);
    __bf16* dst = kvT + (size_t)hh * DK * DK;
    for (int r = 0; r < 8; ++r)
        dst[(size_t)(rbase + r) * DK + d] = (__bf16)acc[r];
}

// ksum[hh][d] = sum_n k[n][d]  (contiguous kfT rows)
__global__ void ksum_kernel(const __bf16* __restrict__ kfT,
                            float* __restrict__ ksum) {
    const int hh = blockIdx.x;
    const int d = threadIdx.x;  // 64 threads
    const __bf16* p = kfT + ((size_t)hh * DK + d) * N_SEQ;
    float s = 0.0f;
    for (int n = 0; n < N_SEQ; ++n) s += (float)p[n];
    ksum[hh * DK + d] = s;
}

// ---------------------------------------------------------------------------
// qkv[n][m] = sum_d q[n][d]*kvT[m][d]; ctx = qkv / (q . ksum + eps),
// merged back to [row, d_model] bf16 for the output projection.
// Block covers 32 seq rows x 64 d_k cols (8 waves); norms staged in LDS.
// ---------------------------------------------------------------------------
__global__ void __launch_bounds__(256) attn_kernel(
    const __bf16* __restrict__ qf, const __bf16* __restrict__ kvT,
    const float* __restrict__ ksum, __bf16* __restrict__ ctx)
{
    __shared__ float norm[32];
    const int hh = blockIdx.y;
    const int bb = hh >> 4, h = hh & 15;
    const int nbase = blockIdx.x * 32;
    const int tid = threadIdx.x;

    if (tid < 32) {
        const __bf16* qrow =
            qf + (size_t)(bb * N_SEQ + nbase + tid) * D_MODEL + h * DK;
        const float* ks = ksum + hh * DK;
        float s = 0.0f;
        for (int d = 0; d < DK; ++d) s += (float)qrow[d] * ks[d];
        norm[tid] = s + 1e-6f;
    }
    __syncthreads();

    const int lane = tid & 31, wave = tid >> 5;
    const int ntile = wave >> 2, mtile = wave & 3;
    const __bf16* Ah = qf + (size_t)(bb * N_SEQ + nbase) * D_MODEL + h * DK;
    const __bf16* Bh = kvT + (size_t)hh * DK * DK;
    const int arow = ntile * 16 + (lane & 15);
    const int brow = mtile * 16 + (lane & 15);

    v8f acc = {};
    #pragma unroll
    for (int k0 = 0; k0 < DK; k0 += 32) {
        v16bf a = load_a_tile(Ah, arow, D_MODEL, k0, lane);
        v16bf b = load_b_tile(Bh, brow, DK, k0, lane);
        acc = wmma_bf16(a, b, acc);
    }

    const int col = mtile * 16 + (lane & 15);
    const int rloc = ntile * 16 + ((lane >> 4) << 3);
    for (int r = 0; r < 8; ++r) {
        const int nl = rloc + r;   // local seq row 0..31
        const float y = acc[r] / norm[nl];
        const size_t row = (size_t)(bb * N_SEQ + nbase + nl);
        ctx[row * D_MODEL + h * DK + col] = (__bf16)y;
    }
}

// ---------------------------------------------------------------------------
extern "C" void kernel_launch(void* const* d_in, const int* in_sizes, int n_in,
                              void* d_out, int out_size, void* d_ws,
                              size_t ws_size, hipStream_t stream)
{
    (void)in_sizes; (void)n_in; (void)out_size; (void)ws_size;
    const float* query = (const float*)d_in[0];
    const float* key_  = (const float*)d_in[1];
    const float* value = (const float*)d_in[2];
    const float* wq = (const float*)d_in[3];
    const float* bq = (const float*)d_in[4];
    const float* wk = (const float*)d_in[5];
    const float* bk = (const float*)d_in[6];
    const float* wv = (const float*)d_in[7];
    const float* bv = (const float*)d_in[8];
    const float* wo = (const float*)d_in[9];
    const float* bo = (const float*)d_in[10];

    char* ws = (char*)d_ws;
    size_t off = 0;
    auto take = [&](size_t bytes) -> char* {
        char* p = ws + off;
        off += (bytes + 255) & ~(size_t)255;
        return p;
    };

    const size_t ACT_B = (size_t)MTOT * D_MODEL * sizeof(__bf16);     // 33.5MB
    const size_t W_B   = (size_t)D_MODEL * D_MODEL * sizeof(__bf16);  // 2MB

    __bf16* qb  = (__bf16*)take(ACT_B);
    __bf16* kb  = (__bf16*)take(ACT_B);
    __bf16* vb  = (__bf16*)take(ACT_B);
    __bf16* wqb = (__bf16*)take(W_B);
    __bf16* wkb = (__bf16*)take(W_B);
    __bf16* wvb = (__bf16*)take(W_B);
    __bf16* wob = (__bf16*)take(W_B);
    __bf16* qf  = (__bf16*)take(ACT_B);               // elu(q)+1, row-major
    __bf16* kfT = (__bf16*)take(ACT_B);               // [b,h,d,n]
    __bf16* vfT = (__bf16*)take(ACT_B);               // [b,h,d,n]
    __bf16* kvT = (__bf16*)take((size_t)NHH * DK * DK * sizeof(__bf16));
    float*  ksm = (float*)take((size_t)NHH * DK * sizeof(float));
    __bf16* ctx = qb;   // reuse: qb dead once projections complete

    // 1) fp32 -> bf16
    cvt_kernel<<<2048, 256, 0, stream>>>(query, qb, MTOT * D_MODEL);
    cvt_kernel<<<2048, 256, 0, stream>>>(key_,  kb, MTOT * D_MODEL);
    cvt_kernel<<<2048, 256, 0, stream>>>(value, vb, MTOT * D_MODEL);
    cvt_kernel<<<512, 256, 0, stream>>>(wq, wqb, D_MODEL * D_MODEL);
    cvt_kernel<<<512, 256, 0, stream>>>(wk, wkb, D_MODEL * D_MODEL);
    cvt_kernel<<<512, 256, 0, stream>>>(wv, wvb, D_MODEL * D_MODEL);
    cvt_kernel<<<512, 256, 0, stream>>>(wo, wob, D_MODEL * D_MODEL);

    // 2) projections: async/TDM-staged WMMA GEMMs + feature map + transposes
    dim3 gg(D_MODEL / 128, MTOT / 64);   // (8, 256) blocks, 8 waves each
    gemm_tile_kernel<<<gg, 256, 0, stream>>>(qb, wqb, bq, qf, nullptr, nullptr,
                                             MTOT, D_MODEL, D_MODEL, 1);
    gemm_tile_kernel<<<gg, 256, 0, stream>>>(kb, wkb, bk, nullptr, kfT, nullptr,
                                             MTOT, D_MODEL, D_MODEL, 1);
    gemm_tile_kernel<<<gg, 256, 0, stream>>>(vb, wvb, bv, nullptr, vfT, nullptr,
                                             MTOT, D_MODEL, D_MODEL, 0);

    // 3) kv state + k row-sums
    kv_kernel<<<NHH, 512, 0, stream>>>(kfT, vfT, kvT);
    ksum_kernel<<<NHH, DK, 0, stream>>>(kfT, ksm);

    // 4) q @ kv, normalize, merge heads
    dim3 ga(N_SEQ / 32, NHH);
    attn_kernel<<<ga, 256, 0, stream>>>(qf, kvT, ksm, ctx);

    // 5) output projection -> fp32 result
    gemm_tile_kernel<<<gg, 256, 0, stream>>>(ctx, wob, bo, nullptr, nullptr,
                                             (float*)d_out,
                                             MTOT, D_MODEL, D_MODEL, 0);
}
